// GraphSpectralFilterLayer_51728586113129
// MI455X (gfx1250) — matrix operand order repaired and checked by reading, not compiled, not run on hardware
//
#include <hip/hip_runtime.h>
#include <math.h>

// ---- CDNA5 WMMA types -------------------------------------------------------
typedef __attribute__((ext_vector_type(16))) __bf16 v16bf;
typedef __attribute__((ext_vector_type(8)))  float  v8f;
typedef __attribute__((ext_vector_type(4)))  float  v4f;

// ---- problem constants ------------------------------------------------------
#define NN      4096          // nodes
#define INF_    512
#define OUTF    128
#define CCH     4
#define ROWS    (CCH * NN)    // 16384
#define INV_ATT (1.0f / 67108864.0f)

// LDS staging geometry for the fused GEMM:
#define KCHUNK  128           // K elements staged per round
#define CSTR    148           // padded column stride (bf16 elems); 74 dwords = 10 mod 64
                              //  -> conflict-free ds b64 reads AND writes

static __device__ __forceinline__ unsigned short f2bf_bits(float f) {
    __bf16 b = (__bf16)f;
    return __builtin_bit_cast(unsigned short, b);
}

// ============================================================================
// 1) global mean partials: 1024 blocks x 256 thr, 64K elems/blk, b128 streams
// ============================================================================
__global__ void k_mean_part(const float* __restrict__ att, float* __restrict__ part) {
    __shared__ float red[256];
    const v4f* att4 = (const v4f*)att;
    size_t base4 = (size_t)blockIdx.x * 16384u;
    v4f s4 = {0.f, 0.f, 0.f, 0.f};
#pragma unroll 8
    for (int i = 0; i < 64; ++i)
        s4 += __builtin_nontemporal_load(att4 + base4 + (size_t)i * 256u + threadIdx.x);
    red[threadIdx.x] = s4[0] + s4[1] + s4[2] + s4[3];
    __syncthreads();
    for (int off = 128; off > 0; off >>= 1) {
        if ((int)threadIdx.x < off) red[threadIdx.x] += red[threadIdx.x + off];
        __syncthreads();
    }
    if (threadIdx.x == 0) part[blockIdx.x] = red[0];
}

// 2) reduce 1024 partials -> scalar total sum (deterministic tree)
__global__ void k_mean_final(const float* __restrict__ part, float* __restrict__ totalSum) {
    __shared__ float red[256];
    float s = part[threadIdx.x] + part[threadIdx.x + 256] +
              part[threadIdx.x + 512] + part[threadIdx.x + 768];
    red[threadIdx.x] = s;
    __syncthreads();
    for (int off = 128; off > 0; off >>= 1) {
        if ((int)threadIdx.x < off) red[threadIdx.x] += red[threadIdx.x + off];
        __syncthreads();
    }
    if (threadIdx.x == 0) totalSum[0] = red[0];
}

// ============================================================================
// 3) column softmax denominators (axis=0). LeakyReLU is identity for kept
//    values (mean>0), exp(0.2*-9e15)==0 for rejected ones, kept a in (0.5,1)
//    needs no max-subtraction:  colsum[k] = sum_m (a>mean ? exp(a) : 0).
//    Each thread owns 4 adjacent columns (b128 loads, wave = 512B/instr).
// ============================================================================
__global__ void k_colsum_part(const float* __restrict__ att,
                              const float* __restrict__ totalSum,
                              float* __restrict__ colpart) {
    const float mean = totalSum[0] * INV_ATT;
    int kb = blockIdx.x & 3;           // 4 k-blocks of 1024 columns
    int mb = blockIdx.x >> 2;          // 0..63 row slabs of 256
    int k4 = kb * 256 + threadIdx.x;   // index in units of 4 columns
    size_t m0 = (size_t)mb * 256u;
    const v4f* att4 = (const v4f*)att; // 1024 v4f per attention row
    v4f s = {0.f, 0.f, 0.f, 0.f};
#pragma unroll 4
    for (int i = 0; i < 256; ++i) {
        v4f a = __builtin_nontemporal_load(att4 + (m0 + i) * 1024u + k4);
#pragma unroll
        for (int e = 0; e < 4; ++e)
            s[e] += (a[e] > mean) ? __expf(a[e]) : 0.f;
    }
    *(v4f*)(colpart + (size_t)mb * NN + (size_t)k4 * 4u) = s;
}

// 4) reduce 64 slabs -> rcs[k] = dropout_scale / colsum[k]
__global__ void k_colsum_final(const float* __restrict__ colpart, float* __restrict__ rcs) {
    int k = blockIdx.x * 256 + threadIdx.x;
    float s = 0.f;
#pragma unroll
    for (int mb = 0; mb < 64; ++mb) s += colpart[(size_t)mb * NN + k];
    rcs[k] = 2.5f / s;                 // 1/(1-0.6) = 2.5 dropout scale folded in
}

// ============================================================================
// 5) h~ = (x @ W^T) * rcs[row] via WMMA bf16, stored transposed: hT[o][m].
//    Folding 2.5/colsum into h~ here (0.5M elems) removes the per-element
//    normalization from the 67M-element fused pass.
// ============================================================================
__global__ void k_h_gemm(const float* __restrict__ x, const float* __restrict__ W,
                         const float* __restrict__ rcs, unsigned short* __restrict__ hT) {
    int wid  = blockIdx.x * 4 + (threadIdx.x >> 5);   // 0..2047
    int lane = threadIdx.x & 31;
    int mtile = wid >> 3;                             // 0..255
    int ntile = wid & 7;                              // 0..7
    int half  = lane >> 4;
    int l16   = lane & 15;
    int mrow  = mtile * 16 + l16;                     // A-matrix row for this lane
    int ocol  = ntile * 16 + l16;                     // B-matrix column

    v8f acc = {0.f, 0.f, 0.f, 0.f, 0.f, 0.f, 0.f, 0.f};

    for (int k0 = 0; k0 < INF_; k0 += 32) {
        // A frag: lanes<16 -> K{0..7,16..23}, lanes>=16 -> K{8..15,24..31}
        int ka = k0 + half * 8;
        union { v4f q[4]; float f[16]; } AV;
        AV.q[0] = *(const v4f*)(x + (size_t)mrow * INF_ + ka);
        AV.q[1] = *(const v4f*)(x + (size_t)mrow * INF_ + ka + 4);
        AV.q[2] = *(const v4f*)(x + (size_t)mrow * INF_ + ka + 16);
        AV.q[3] = *(const v4f*)(x + (size_t)mrow * INF_ + ka + 20);
        v16bf A;
#pragma unroll
        for (int e = 0; e < 16; ++e) A[e] = (__bf16)AV.f[e];

        // B frag: B[k][o] = W[o][k]; lanes<16 -> K{0..7,16..23}, lanes>=16 -> K{8..15,24..31}
        int kb = k0 + half * 8;
        union { v4f q[4]; float f[16]; } BV;
        BV.q[0] = *(const v4f*)(W + (size_t)ocol * INF_ + kb);
        BV.q[1] = *(const v4f*)(W + (size_t)ocol * INF_ + kb + 4);
        BV.q[2] = *(const v4f*)(W + (size_t)ocol * INF_ + kb + 16);
        BV.q[3] = *(const v4f*)(W + (size_t)ocol * INF_ + kb + 20);
        v16bf B;
#pragma unroll
        for (int e = 0; e < 16; ++e) B[e] = (__bf16)BV.f[e];

        acc = __builtin_amdgcn_wmma_f32_16x16x32_bf16(false, A, false, B,
                                                      (short)0, acc, false, false);
    }

    // D layout: VGPR v -> M = v (+8 for upper half-wave), N = l16.
    // Scale by rcs[m] (the big GEMM's per-K normalization, folded into B).
    union { v4f q[2]; float f[8]; } R;
    R.q[0] = *(const v4f*)(rcs + mtile * 16 + half * 8);
    R.q[1] = *(const v4f*)(rcs + mtile * 16 + half * 8 + 4);
#pragma unroll
    for (int v = 0; v < 8; ++v) {
        int m = mtile * 16 + v + half * 8;
        hT[(size_t)ocol * NN + m] = f2bf_bits(acc[v] * R.f[v]);
    }
}

// ============================================================================
// 6) fused threshold/exp/dropout + (Q @ h~) + permuted scatter.
//    One wave per 16-row M-tile, 4 waves/block. h~ K-chunks staged in LDS
//    (conflict-free padded layout), shared by the block: B-operand L2 traffic
//    drops from ~8 GB to ~256 MB. A operand is pure NT streaming from HBM.
// ============================================================================
__global__ void __launch_bounds__(128)
k_attn_gemm(const float* __restrict__ att,
            const unsigned char* __restrict__ mask,
            const float* __restrict__ totalSum,
            const unsigned short* __restrict__ hT,
            float* __restrict__ out) {
    __shared__ unsigned short sB[OUTF * CSTR];        // 128*148*2B = 37 KB

    const float mean = totalSum[0] * INV_ATT;
    int tid  = threadIdx.x;
    int wv   = tid >> 5;                              // wave in block, 0..3
    int lane = tid & 31;
    int half = lane >> 4;
    int l16  = lane & 15;
    int wid  = blockIdx.x * 4 + wv;                   // m-tile id, 0..1023
    size_t mrow = (size_t)wid * 16u + (size_t)l16;    // attention row this lane packs

    v8f acc[8];
#pragma unroll
    for (int t = 0; t < 8; ++t) acc[t] = (v8f){0.f,0.f,0.f,0.f,0.f,0.f,0.f,0.f};

    const float*         arow = att  + mrow * (size_t)NN;
    const unsigned char* mrw  = mask + mrow * (size_t)NN;

    for (int kc = 0; kc < NN; kc += KCHUNK) {
        __syncthreads();   // previous chunk fully consumed before overwrite
        // ---- stage hT[:, kc:kc+128] -> LDS. Per instr: one column per wave,
        // lanes cover 256 contiguous bytes (coalesced global b64 reads; LDS
        // write banks = 2*lane -> conflict-free).
#pragma unroll 8
        for (int i = 0; i < 32; ++i) {
            int o = i * 4 + wv;
            unsigned long long v =
                *(const unsigned long long*)(hT + (size_t)o * NN + kc + lane * 4);
            *(unsigned long long*)(sB + o * CSTR + lane * 4) = v;
        }
        __syncthreads();

        // ---- 4 K-steps of 32 against the staged chunk
#pragma unroll
        for (int s = 0; s < 4; ++s) {
            int k0 = kc + s * 32;
            int ka = k0 + half * 8;
            union { v4f q[4]; float f[16]; } AV;
            AV.q[0] = __builtin_nontemporal_load((const v4f*)(arow + ka));
            AV.q[1] = __builtin_nontemporal_load((const v4f*)(arow + ka + 4));
            AV.q[2] = __builtin_nontemporal_load((const v4f*)(arow + ka + 16));
            AV.q[3] = __builtin_nontemporal_load((const v4f*)(arow + ka + 20));
            union { unsigned long long u[2]; unsigned char b[16]; } MV;
            MV.u[0] = __builtin_nontemporal_load((const unsigned long long*)(mrw + ka));
            MV.u[1] = __builtin_nontemporal_load((const unsigned long long*)(mrw + ka + 16));

            v16bf A;
#pragma unroll
            for (int e = 0; e < 16; ++e) {
                float a = AV.f[e];
                float p = ((a > mean) && MV.b[e]) ? __expf(a) : 0.f;
                A[e] = (__bf16)p;
            }

#pragma unroll
            for (int t = 0; t < 8; ++t) {
                // B frag from LDS: lanes<16 -> K{0..7,16..23}, lanes>=16 -> K{8..15,24..31}
                const unsigned short* cb = sB + (t * 16 + l16) * CSTR + s * 32 + half * 8;
                union { unsigned long long u[4]; v16bf v; } B;
                B.u[0] = *(const unsigned long long*)(cb + 0);
                B.u[1] = *(const unsigned long long*)(cb + 4);
                B.u[2] = *(const unsigned long long*)(cb + 16);
                B.u[3] = *(const unsigned long long*)(cb + 20);
                acc[t] = __builtin_amdgcn_wmma_f32_16x16x32_bf16(false, A, false, B.v,
                                                                 (short)0, acc[t],
                                                                 false, false);
            }
        }
    }

    // scatter: h_prime row m=(c,n) -> out[n*512 + c*128 + f]
#pragma unroll
    for (int t = 0; t < 8; ++t) {
        int f = t * 16 + l16;
#pragma unroll
        for (int v = 0; v < 8; ++v) {
            int m = wid * 16 + v + half * 8;
            int c = m >> 12;
            int n = m & (NN - 1);
            out[(size_t)n * 512u + (size_t)(c * OUTF + f)] = acc[t][v];
        }
    }
}

// ============================================================================
extern "C" void kernel_launch(void* const* d_in, const int* in_sizes, int n_in,
                              void* d_out, int out_size, void* d_ws, size_t ws_size,
                              hipStream_t stream) {
    const float*         x    = (const float*)d_in[0];
    const float*         att  = (const float*)d_in[1];
    const float*         W    = (const float*)d_in[2];
    const unsigned char* mask = (const unsigned char*)d_in[3];
    float* out = (float*)d_out;

    // workspace layout (floats): [0..1023] mean partials | [1024] total sum |
    // [2048..6143] rcs | [8192..] colpart (1MB) reused afterwards as hT (bf16)
    float* ws        = (float*)d_ws;
    float* meanPart  = ws;
    float* totalSum  = ws + 1024;
    float* rcs       = ws + 2048;
    float* colpart   = ws + 8192;
    unsigned short* hT = (unsigned short*)(ws + 8192);  // reused after k_colsum_final

    k_mean_part   <<<1024, 256, 0, stream>>>(att, meanPart);
    k_mean_final  <<<1,    256, 0, stream>>>(meanPart, totalSum);
    k_colsum_part <<<256,  256, 0, stream>>>(att, totalSum, colpart);
    k_colsum_final<<<16,   256, 0, stream>>>(colpart, rcs);
    k_h_gemm      <<<512,  128, 0, stream>>>(x, W, rcs, hT);
    k_attn_gemm   <<<256,  128, 0, stream>>>(att, mask, totalSum, hT, out);
}
